// QGNN2Layer_28217935135269
// MI455X (gfx1250) — matrix operand order repaired and checked by reading, not compiled, not run on hardware
//
#include <hip/hip_runtime.h>
#include <hip/hip_bf16.h>

// ---------------------------------------------------------------------------
// GNN layer for MI455X (gfx1250, wave32, WMMA).
//   edge MLP : [E,320] @ [320,128] -> SiLU -> @ [128,64] -> SiLU
//   scatter  : atomic add into aggr[receiver]
//   node MLP : [N,192] @ [192,128] -> SiLU -> @ [128,128]
// Edge kernel is block-cooperative: 64 edges/block staged to LDS once,
// weights held in registers per wave (one N-tile per wave) to cut L2
// weight traffic ~3x vs wave-private tiles.
// ---------------------------------------------------------------------------

typedef __attribute__((ext_vector_type(16))) __bf16 v16bf;
typedef __attribute__((ext_vector_type(8)))  __bf16 v8bf;
typedef __attribute__((ext_vector_type(8)))  float  v8f;
typedef __attribute__((ext_vector_type(4)))  float  v4f;

#define WPB 8           // waves per block (256 threads)
#define N_NODES 50000
#define N_EDGES 800000

#define A_STRIDE 328    // 320 + 8 pad (bf16) -> 656B rows, bank-spread
#define H_STRIDE 136    // 128 + 8 pad

// ---- helpers --------------------------------------------------------------

// WMMA A fragment from a global fp32 row segment. Lane (l<16) holds
// K = base+0..7 / base+16..23; lane (l>=16) holds base+8..15 / base+24..31.
// Caller folds (lane>=16 ? 8 : 0) into p, so we read p[0..7], p[16..23].
__device__ __forceinline__ v16bf a_from_global(const float* __restrict__ p) {
    const v4f* q = (const v4f*)p;
    v4f c0 = q[0], c1 = q[1], c2 = q[4], c3 = q[5];
    v16bf a;
    a[0]=(__bf16)c0[0]; a[1]=(__bf16)c0[1]; a[2]=(__bf16)c0[2]; a[3]=(__bf16)c0[3];
    a[4]=(__bf16)c1[0]; a[5]=(__bf16)c1[1]; a[6]=(__bf16)c1[2]; a[7]=(__bf16)c1[3];
    a[8]=(__bf16)c2[0]; a[9]=(__bf16)c2[1]; a[10]=(__bf16)c2[2]; a[11]=(__bf16)c2[3];
    a[12]=(__bf16)c3[0]; a[13]=(__bf16)c3[1]; a[14]=(__bf16)c3[2]; a[15]=(__bf16)c3[3];
    return a;
}

// Same chunk pattern, bf16 source in LDS (two ds_load_b128).
__device__ __forceinline__ v16bf a_from_lds(const __bf16* p) {
    const v8bf* q = (const v8bf*)p;
    v8bf c0 = q[0];
    v8bf c1 = q[2];   // +16 elements
    v16bf a;
#pragma unroll
    for (int i = 0; i < 8; ++i) { a[i] = c0[i]; a[i + 8] = c1[i]; }
    return a;
}

__device__ __forceinline__ float silu(float x) {
    return x / (1.0f + __expf(-x));
}

#define WMMA_BF16(A, B, C) \
    __builtin_amdgcn_wmma_f32_16x16x32_bf16(false, (A), false, (B), (short)0, (C), false, false)

// ---- weight prepack: fp32 [K,N] row-major -> per-lane WMMA B layout -------
// Packed as [ntile][kstep][lane][16 bf16]; lane holds col nt*16+(lane&15),
// K = kk*32 + (lane<16?0:8) + {0..7, 16..23} (32x16 bf16 B fragment).
__global__ void pack_b_bf16(const float* __restrict__ W, __bf16* __restrict__ dst,
                            int K, int N) {
    int t = blockIdx.x * blockDim.x + threadIdx.x;
    int total = K * N;
    if (t >= total) return;
    int KS   = K >> 5;
    int i    = t & 15;
    int lane = (t >> 4) & 31;
    int kk   = (t >> 9) % KS;
    int nt   = t / (KS << 9);
    int k = kk * 32 + ((lane & 16) ? 8 : 0) + (i & 7) + ((i & 8) ? 16 : 0);
    int n = nt * 16 + (lane & 15);
    dst[t] = (__bf16)W[(size_t)k * N + n];
}

__global__ void zero_f32(float* __restrict__ p, int n) {
    int i = blockIdx.x * blockDim.x + threadIdx.x;
    if (i < n) p[i] = 0.0f;
}

// ---- edge kernel: 64 edges per block, cooperative -------------------------
__global__ __launch_bounds__(256) void edge_mlp_kernel(
    const float* __restrict__ x_nodes, const float* __restrict__ x_edges,
    const int* __restrict__ snd_idx,   const int* __restrict__ rcv_idx,
    const float* __restrict__ eb1,     const float* __restrict__ eb2,
    const __bf16* __restrict__ w1p,    const __bf16* __restrict__ w2p,
    float* __restrict__ edges_out,     float* __restrict__ aggr) {

    __shared__ int sidx[64];
    __shared__ int ridx[64];
    __shared__ __align__(16) __bf16 Asm[64][A_STRIDE];  // 41984 B
    __shared__ __align__(16) __bf16 hsm[64][H_STRIDE];  // 17408 B

    const int tid  = threadIdx.x;
    const int lane = tid & 31;
    const int wave = tid >> 5;
    const int m    = lane & 15;
    const int hi   = lane >> 4;
    const int e0   = blockIdx.x * 64;

    // ---- stage edge indices ----
    if (tid < 64) {
        sidx[tid] = snd_idx[e0 + tid];
        ridx[tid] = rcv_idx[e0 + tid];
    }
    __syncthreads();

    // ---- stage A = concat(x_s, x_r, x_e) as bf16 [64][320] in LDS ----
    // 64 rows * 80 float4 chunks, 256 threads -> 20 chunks/thread
    for (int j = tid; j < 64 * 80; j += 256) {
        int row = j / 80;
        int cf  = (j % 80) * 4;
        const float* src;
        int f;
        if (cf < 128)      { src = x_nodes + (size_t)sidx[row] * 128; f = cf; }
        else if (cf < 256) { src = x_nodes + (size_t)ridx[row] * 128; f = cf - 128; }
        else               { src = x_edges + (size_t)(e0 + row) * 64; f = cf - 256; }
        v4f v = *(const v4f*)(src + f);
        __bf16* d = &Asm[row][cf];
        d[0] = (__bf16)v[0]; d[1] = (__bf16)v[1];
        d[2] = (__bf16)v[2]; d[3] = (__bf16)v[3];
    }
    __syncthreads();

    const v16bf* w1v = (const v16bf*)w1p;
    const v16bf* w2v = (const v16bf*)w2p;

    // ---- GEMM1: [64,320] x [320,128]; wave owns N-tile nt=wave ----
    // B fragments live in registers across all 4 M-tiles.
    v16bf b1[10];
#pragma unroll
    for (int kk = 0; kk < 10; ++kk)
        b1[kk] = w1v[(wave * 10 + kk) * 32 + lane];

    v8f acc[4];
#pragma unroll
    for (int i = 0; i < 4; ++i) { v8f z{}; acc[i] = z; }

#pragma unroll
    for (int mt = 0; mt < 4; ++mt) {
#pragma unroll
        for (int kk = 0; kk < 10; ++kk) {
            v16bf a = a_from_lds(&Asm[mt * 16 + m][kk * 32 + hi * 8]);
            acc[mt] = WMMA_BF16(a, b1[kk], acc[mt]);
        }
    }

    // ---- epilogue 1: +bias, SiLU -> hsm ----
    {
        int c = wave * 16 + m;
        float b = eb1[c];
#pragma unroll
        for (int mt = 0; mt < 4; ++mt) {
#pragma unroll
            for (int r = 0; r < 8; ++r) {
                float h = silu(acc[mt][r] + b);
                hsm[mt * 16 + r + hi * 8][c] = (__bf16)h;
            }
        }
    }
    __syncthreads();

    // ---- GEMM2: [64,128] x [128,64]; wave -> (M-tile, N-half) ----
    const int mt2 = wave & 3;
    const int nh  = wave >> 2;

    v16bf b2[2][4];
#pragma unroll
    for (int j = 0; j < 2; ++j)
#pragma unroll
        for (int kk = 0; kk < 4; ++kk)
            b2[j][kk] = w2v[((nh * 2 + j) * 4 + kk) * 32 + lane];

    v8f acc2[2];
#pragma unroll
    for (int i = 0; i < 2; ++i) { v8f z{}; acc2[i] = z; }

#pragma unroll
    for (int kk = 0; kk < 4; ++kk) {
        v16bf a = a_from_lds(&hsm[mt2 * 16 + m][kk * 32 + hi * 8]);
#pragma unroll
        for (int j = 0; j < 2; ++j)
            acc2[j] = WMMA_BF16(a, b2[j][kk], acc2[j]);
    }

    // ---- epilogue 2: +bias, SiLU, store edge output, scatter-add ----
#pragma unroll
    for (int r = 0; r < 8; ++r) {
        int row = mt2 * 16 + r + hi * 8;
        int ee  = e0 + row;
        int rv  = ridx[row];
#pragma unroll
        for (int j = 0; j < 2; ++j) {
            int c = (nh * 2 + j) * 16 + m;
            float v = silu(acc2[j][r] + eb2[c]);
            edges_out[(size_t)ee * 64 + c] = v;
            unsafeAtomicAdd(&aggr[(size_t)rv * 64 + c], v);
        }
    }
}

// ---- node kernel: 16 nodes per wave ---------------------------------------
__global__ __launch_bounds__(256) void node_mlp_kernel(
    const float* __restrict__ x_nodes, const float* __restrict__ aggr,
    const float* __restrict__ nb1,     const float* __restrict__ nb2,
    const __bf16* __restrict__ w1p,    const __bf16* __restrict__ w2p,
    float* __restrict__ nodes_out,     int n_tiles) {

    __shared__ __align__(16) __bf16 hsm[WPB][16][128];

    const int lane = threadIdx.x & 31;
    const int wave = threadIdx.x >> 5;
    const int tile = blockIdx.x * WPB + wave;
    if (tile >= n_tiles) return;                  // wave-uniform exit
    const int n0 = tile * 16;
    const int m  = lane & 15;
    const int hi = lane >> 4;
    const int nd = n0 + m;

    const float* nrow = x_nodes + (size_t)nd * 128;
    const float* arow = aggr    + (size_t)nd * 64;

    const v16bf* w1v = (const v16bf*)w1p;
    const v16bf* w2v = (const v16bf*)w2p;

    // ---- GEMM1: [16,192] x [192,128], K in 6 steps ----
    v8f acc[8];
#pragma unroll
    for (int i = 0; i < 8; ++i) { v8f z{}; acc[i] = z; }

#pragma unroll
    for (int kk = 0; kk < 6; ++kk) {
        const float* src = (kk < 4) ? (nrow + kk * 32) : (arow + (kk - 4) * 32);
        v16bf a = a_from_global(src + hi * 8);
#pragma unroll
        for (int nt = 0; nt < 8; ++nt)
            acc[nt] = WMMA_BF16(a, w1v[(nt * 6 + kk) * 32 + lane], acc[nt]);
    }

    // ---- epilogue 1: +bias, SiLU -> LDS ----
#pragma unroll
    for (int nt = 0; nt < 8; ++nt) {
        int c = nt * 16 + m;
        float b = nb1[c];
#pragma unroll
        for (int r = 0; r < 8; ++r) {
            float h = silu(acc[nt][r] + b);
            hsm[wave][r + hi * 8][c] = (__bf16)h;
        }
    }

    // ---- GEMM2: [16,128] x [128,128], K in 4 steps ----
    v8f acc2[8];
#pragma unroll
    for (int i = 0; i < 8; ++i) { v8f z{}; acc2[i] = z; }

#pragma unroll
    for (int kk = 0; kk < 4; ++kk) {
        v16bf a = a_from_lds(&hsm[wave][m][kk * 32 + hi * 8]);
#pragma unroll
        for (int nt = 0; nt < 8; ++nt)
            acc2[nt] = WMMA_BF16(a, w2v[(nt * 4 + kk) * 32 + lane], acc2[nt]);
    }

    // ---- epilogue 2: +bias (no activation), store ----
#pragma unroll
    for (int nt = 0; nt < 8; ++nt) {
        int c = nt * 16 + m;
        float b = nb2[c];
#pragma unroll
        for (int r = 0; r < 8; ++r) {
            int row = r + hi * 8;
            nodes_out[(size_t)(n0 + row) * 128 + c] = acc2[nt][r] + b;
        }
    }
}

// ---------------------------------------------------------------------------

extern "C" void kernel_launch(void* const* d_in, const int* in_sizes, int n_in,
                              void* d_out, int out_size, void* d_ws, size_t ws_size,
                              hipStream_t stream) {
    const float* x_nodes = (const float*)d_in[0];
    const float* x_edges = (const float*)d_in[1];
    const int*   eidx    = (const int*)d_in[2];   // [2, E] int32
    const float* eW1 = (const float*)d_in[3];
    const float* eb1 = (const float*)d_in[4];
    const float* eW2 = (const float*)d_in[5];
    const float* eb2 = (const float*)d_in[6];
    const float* nW1 = (const float*)d_in[7];
    const float* nb1 = (const float*)d_in[8];
    const float* nW2 = (const float*)d_in[9];
    const float* nb2 = (const float*)d_in[10];

    const int* snd = eidx;
    const int* rcv = eidx + N_EDGES;

    // outputs: x_nodes_out [N,128] then x_edges_out [E,64]
    float* nodes_out = (float*)d_out;
    float* edges_out = nodes_out + (size_t)N_NODES * 128;

    // workspace layout (32B-aligned regions)
    char* ws = (char*)d_ws;
    float*  aggr = (float*)ws;                                  // N*64 f32 = 12.8 MB
    __bf16* w1p  = (__bf16*)(ws + 12800000);                    // 320*128 bf16
    __bf16* w2p  = (__bf16*)(ws + 12800000 + 81920);            // 128*64
    __bf16* nw1p = (__bf16*)(ws + 12800000 + 81920 + 16384);    // 192*128
    __bf16* nw2p = (__bf16*)(ws + 12800000 + 81920 + 16384 + 49152); // 128*128

    // 1) zero the aggregation buffer (ws is not re-poisoned between replays)
    zero_f32<<<(N_NODES * 64 + 255) / 256, 256, 0, stream>>>(aggr, N_NODES * 64);

    // 2) prepack weights to bf16 WMMA-B layout
    pack_b_bf16<<<(320 * 128 + 255) / 256, 256, 0, stream>>>(eW1, w1p, 320, 128);
    pack_b_bf16<<<(128 * 64  + 255) / 256, 256, 0, stream>>>(eW2, w2p, 128, 64);
    pack_b_bf16<<<(192 * 128 + 255) / 256, 256, 0, stream>>>(nW1, nw1p, 192, 128);
    pack_b_bf16<<<(128 * 128 + 255) / 256, 256, 0, stream>>>(nW2, nw2p, 128, 128);

    // 3) edge MLP + scatter-add: 800000/64 = 12500 cooperative blocks
    edge_mlp_kernel<<<N_EDGES / 64, 256, 0, stream>>>(
        x_nodes, x_edges, snd, rcv, eb1, eb2, w1p, w2p, edges_out, aggr);

    // 4) node MLP: 50000/16 = 3125 tiles
    int n_tiles = N_NODES / 16;
    node_mlp_kernel<<<(n_tiles + WPB - 1) / WPB, 256, 0, stream>>>(
        x_nodes, aggr, nb1, nb2, nw1p, nw2p, nodes_out, n_tiles);
}
